// WeightedCommonNeighborsPredictor_84310208020585
// MI455X (gfx1250) — matrix operand order repaired and checked by reading, not compiled, not run on hardware
//
#include <hip/hip_runtime.h>
#include <stdint.h>

#define DCH 128
#define COS_EPS 1e-8f
#define DEG_EPS 1e-6f

typedef __attribute__((ext_vector_type(16))) __bf16 bf16x16;
typedef __attribute__((ext_vector_type(8)))  float  f32x8;

// ---------------------------------------------------------------- degrees
__global__ void k_deg(const int* __restrict__ row, float* __restrict__ deg, int M) {
    int i = blockIdx.x * blockDim.x + threadIdx.x;
    int stride = gridDim.x * blockDim.x;
    for (; i < M; i += stride) atomicAdd(&deg[row[i]], 1.0f);
}

// ------------------------------------------------- row-sorted COO aggregation
// 128 threads = 1 channel each; runs of equal row accumulate in registers,
// flush with one atomic per (run, channel).
#define AGG_CHUNK 512
__global__ __launch_bounds__(128) void k_agg(const float* __restrict__ x,
                                             const int* __restrict__ row,
                                             const int* __restrict__ col,
                                             float* __restrict__ agg, int M) {
    const int c = threadIdx.x;
    long start = (long)blockIdx.x * AGG_CHUNK;
    long end   = start + AGG_CHUNK; if (end > M) end = M;
    float acc = 0.0f; int cur = -1;
    for (long e = start; e < end; ++e) {
        int r  = row[e];
        int cl = col[e];
        if (r != cur) {
            if (cur >= 0) atomicAdd(&agg[(long)cur * DCH + c], acc);
            cur = r; acc = 0.0f;
        }
        acc += x[(long)cl * DCH + c];
    }
    if (cur >= 0) atomicAdd(&agg[(long)cur * DCH + c], acc);
}

// -------------------------------------- h0 = x + agg/deg, emit bf16 for WMMA
__global__ void k_combine(const float* __restrict__ x, const float* __restrict__ agg,
                          const float* __restrict__ deg, __bf16* __restrict__ x0b,
                          long total) {
    long n4 = total >> 2;
    long stride = (long)gridDim.x * blockDim.x;
    for (long i4 = (long)blockIdx.x * blockDim.x + threadIdx.x; i4 < n4; i4 += stride) {
        long base = i4 << 2;
        long rowi = base >> 7;                 // /128
        float inv = 1.0f / (deg[rowi] + DEG_EPS);
        float4 xv = ((const float4*)x)[i4];
        float4 av = ((const float4*)agg)[i4];
        x0b[base + 0] = (__bf16)(xv.x + av.x * inv);
        x0b[base + 1] = (__bf16)(xv.y + av.y * inv);
        x0b[base + 2] = (__bf16)(xv.z + av.z * inv);
        x0b[base + 3] = (__bf16)(xv.w + av.w * inv);
    }
}

// --------------------------------------------- fused 3-layer MLP via WMMA
// Block = 256 threads = 8 waves; block handles 32 rows (two 16-row M-tiles),
// wave handles one 16-column N-tile across both M-tiles. Per layer the
// weights are transposed into LDS as bf16 [n][k]; each B fragment is loaded
// once and reused for both M-tiles -> 8 wmma_f32_16x16x32_bf16 per layer.
#define MROWS 32
__global__ __launch_bounds__(256) void k_mlp(const __bf16* __restrict__ x0b,
                                             const float* __restrict__ W1, const float* __restrict__ b1,
                                             const float* __restrict__ W2, const float* __restrict__ b2,
                                             const float* __restrict__ W3, const float* __restrict__ b3,
                                             float* __restrict__ hout, int N) {
    __shared__ __bf16 sW[DCH * DCH];        // transposed: sW[n*128 + k]
    __shared__ __bf16 sA[MROWS * DCH];      // activations, row-major [m][k]
    __shared__ float  sBias[3 * DCH];

    const int tid  = threadIdx.x;
    const int wave = tid >> 5;              // output column tile (0..7)
    const int lane = tid & 31;
    const int row0 = blockIdx.x * MROWS;
    const int hi   = (lane >> 4) & 1;       // lane half
    const int colg = (wave << 4) + (lane & 15);

    // preload activation tile (already bf16) and all three biases
    {
        const uint32_t* src = (const uint32_t*)(x0b + (long)row0 * DCH);
        uint32_t* dst = (uint32_t*)sA;
        for (int i = tid; i < MROWS * DCH / 2; i += 256) dst[i] = src[i];
        for (int i = tid; i < 3 * DCH; i += 256)
            sBias[i] = (i < DCH) ? b1[i] : (i < 2 * DCH) ? b2[i - DCH] : b3[i - 2 * DCH];
    }

    for (int l = 0; l < 3; ++l) {
        const float* W = (l == 0) ? W1 : (l == 1) ? W2 : W3;
        // load weights, convert to bf16, transpose into LDS
        for (int i = tid; i < DCH * DCH; i += 256) {
            int k = i >> 7, n = i & (DCH - 1);
            sW[n * DCH + k] = (__bf16)W[i];
        }
        __syncthreads();                    // sW ready; prior sA writes visible

        f32x8 acc0 = {};
        f32x8 acc1 = {};
        const int m = lane & 15;
        #pragma unroll
        for (int kt = 0; kt < 4; ++kt) {
            union { bf16x16 v; uint32_t u[8]; } a0, a1, bfrag;
            #pragma unroll
            for (int r = 0; r < 8; ++r) {
                // B 32x16 bf16 layout: V0-3 -> K pairs {0..7}/{8..15}, V4-7 -> +16
                int kb = kt * 32 + ((r >> 2) << 4) + (hi << 3) + ((r & 3) << 1);
                bfrag.u[r] = *(const uint32_t*)&sW[colg * DCH + kb];
                // A 16x32 bf16 layout: V0-3 -> K pairs {0..7}/{8..15}, V4-7 -> +16
                int ka = kt * 32 + ((r >> 2) << 4) + (hi << 3) + ((r & 3) << 1);
                a0.u[r] = *(const uint32_t*)&sA[m * DCH + ka];
                a1.u[r] = *(const uint32_t*)&sA[(16 + m) * DCH + ka];
            }
            acc0 = __builtin_amdgcn_wmma_f32_16x16x32_bf16(false, a0.v, false, bfrag.v,
                                                           (short)0, acc0, false, false);
            acc1 = __builtin_amdgcn_wmma_f32_16x16x32_bf16(false, a1.v, false, bfrag.v,
                                                           (short)0, acc1, false, false);
        }
        __syncthreads();                    // all sA / sW reads complete

        float bias = sBias[l * DCH + colg];
        if (l < 2) {
            #pragma unroll
            for (int r = 0; r < 8; ++r) {
                int m_out = r + (hi << 3);                     // C/D layout
                float f0 = acc0[r] + bias;
                float f1 = acc1[r] + bias;
                f0 = f0 > 0.0f ? f0 : 0.0f;                    // ReLU
                f1 = f1 > 0.0f ? f1 : 0.0f;
                sA[m_out * DCH + colg]        = (__bf16)f0;
                sA[(16 + m_out) * DCH + colg] = (__bf16)f1;
            }
            // made visible by the __syncthreads after next layer's weight load
        } else {
            #pragma unroll
            for (int r = 0; r < 8; ++r) {
                int m_out = r + (hi << 3);
                hout[(long)(row0 + m_out) * DCH + colg]      = acc0[r] + bias;
                hout[(long)(row0 + 16 + m_out) * DCH + colg] = acc1[r] + bias;
            }
        }
    }
}

// ------------------------------------------------------- per-row L2 norms
__global__ __launch_bounds__(256) void k_norm(const float* __restrict__ h,
                                              float* __restrict__ nrm, int N) {
    int wave = threadIdx.x >> 5, lane = threadIdx.x & 31;
    int rowi = blockIdx.x * 8 + wave;
    if (rowi >= N) return;
    float4 p = ((const float4*)(h + (long)rowi * DCH))[lane];
    float s = p.x * p.x + p.y * p.y + p.z * p.z + p.w * p.w;
    for (int off = 16; off; off >>= 1) s += __shfl_xor(s, off, 32);
    if (lane == 0) nrm[rowi] = sqrtf(s);
}

// ------------------------------------- cosine pair weights + segment sum
__global__ __launch_bounds__(256) void k_pairs(const float* __restrict__ h,
                                               const float* __restrict__ nrm,
                                               const int* __restrict__ edges,
                                               const int* __restrict__ cn_edge,
                                               const int* __restrict__ cn_node,
                                               const unsigned char* __restrict__ valid,
                                               float* __restrict__ scores, int E, int P) {
    int wave = threadIdx.x >> 5, lane = threadIdx.x & 31;
    int p = blockIdx.x * 8 + wave;
    if (p >= P) return;
    int e = cn_edge[p];
    int u = edges[e], v = edges[E + e], c = cn_node[p];
    float4 a  = ((const float4*)(h + (long)u * DCH))[lane];
    float4 b  = ((const float4*)(h + (long)v * DCH))[lane];
    float4 cc = ((const float4*)(h + (long)c * DCH))[lane];
    float duc = a.x * cc.x + a.y * cc.y + a.z * cc.z + a.w * cc.w;
    float dvc = b.x * cc.x + b.y * cc.y + b.z * cc.z + b.w * cc.w;
    for (int off = 16; off; off >>= 1) {
        duc += __shfl_xor(duc, off, 32);
        dvc += __shfl_xor(dvc, off, 32);
    }
    if (lane == 0 && valid[p]) {
        float nu = nrm[u], nv = nrm[v], nc = nrm[c];
        float w = (duc / fmaxf(nu * nc, COS_EPS)) * (dvc / fmaxf(nv * nc, COS_EPS));
        atomicAdd(&scores[e], w);
    }
}

// ------------------------------------------------------------------ sigmoid
__global__ void k_sigmoid(const float* __restrict__ scores, float* __restrict__ out, int E) {
    int i = blockIdx.x * blockDim.x + threadIdx.x;
    if (i < E) out[i] = 1.0f / (1.0f + expf(-scores[i]));
}

extern "C" void kernel_launch(void* const* d_in, const int* in_sizes, int n_in,
                              void* d_out, int out_size, void* d_ws, size_t ws_size,
                              hipStream_t stream) {
    const float* x        = (const float*)d_in[0];
    const int*   adj_row  = (const int*)d_in[1];
    const int*   adj_col  = (const int*)d_in[2];
    const int*   edges    = (const int*)d_in[3];
    const int*   cn_e     = (const int*)d_in[4];
    const int*   cn_n     = (const int*)d_in[5];
    const unsigned char* cn_v = (const unsigned char*)d_in[6];
    const float* W1 = (const float*)d_in[7];
    const float* b1 = (const float*)d_in[8];
    const float* W2 = (const float*)d_in[9];
    const float* b2 = (const float*)d_in[10];
    const float* W3 = (const float*)d_in[11];
    const float* b3 = (const float*)d_in[12];

    const int N = in_sizes[0] / DCH;
    const int M = in_sizes[1];
    const int E = in_sizes[3] / 2;
    const int P = in_sizes[4];

    char* ws = (char*)d_ws;
    size_t off = 0;
    auto alloc = [&](size_t bytes) {
        size_t o = off;
        off = (off + bytes + 255) & ~(size_t)255;
        return o;
    };
    float*  deg    = (float*)(ws + alloc((size_t)N * 4));
    float*  agg    = (float*)(ws + alloc((size_t)N * DCH * 4)); // reused as MLP output h
    __bf16* x0b    = (__bf16*)(ws + alloc((size_t)N * DCH * 2));
    float*  nrm    = (float*)(ws + alloc((size_t)N * 4));
    float*  scores = (float*)(ws + alloc((size_t)E * 4));
    (void)ws_size; (void)n_in; (void)out_size;

    hipMemsetAsync(deg, 0, (size_t)N * 4, stream);
    hipMemsetAsync(agg, 0, (size_t)N * DCH * 4, stream);
    hipMemsetAsync(scores, 0, (size_t)E * 4, stream);

    k_deg<<<1024, 256, 0, stream>>>(adj_row, deg, M);
    k_agg<<<(M + AGG_CHUNK - 1) / AGG_CHUNK, 128, 0, stream>>>(x, adj_row, adj_col, agg, M);

    long total = (long)N * DCH;
    long n4 = total >> 2;
    int cblocks = (int)((n4 + 255) / 256); if (cblocks > 32768) cblocks = 32768;
    k_combine<<<cblocks, 256, 0, stream>>>(x, agg, deg, x0b, total);

    k_mlp<<<N / MROWS, 256, 0, stream>>>(x0b, W1, b1, W2, b2, W3, b3, agg, N);
    k_norm<<<(N + 7) / 8, 256, 0, stream>>>(agg, nrm, N);
    k_pairs<<<(P + 7) / 8, 256, 0, stream>>>(agg, nrm, edges, cn_e, cn_n, cn_v, scores, E, P);
    k_sigmoid<<<(E + 255) / 256, 256, 0, stream>>>(scores, (float*)d_out, E);
}
